// VectorNetExport_88381837017692
// MI455X (gfx1250) — matrix operand order repaired and checked by reading, not compiled, not run on hardware
//
#include <hip/hip_runtime.h>
#include <hip/hip_bf16.h>
#include <math.h>

// ---------------------------------------------------------------------------
// VectorNet forward for MI455X (gfx1250, wave32, WMMA bf16).
//  * All heavy matmuls use v_wmma_f32_16x16x32_bf16 (f32 accumulate).
//  * Activations stored bf16 in workspace; LayerNorm math in f32.
//  * Scatter-max done via a CSR (built once per launch) + gather seg-max:
//    deterministic (max is order-independent) and atomic-contention free.
//  * Attention: only gf[0] is consumed by the reference, so we do
//    single-query attention (q0 vs all 8192 keys).
// ---------------------------------------------------------------------------

typedef __bf16 bf16;
typedef __attribute__((ext_vector_type(16))) __bf16 v16bf;
typedef __attribute__((ext_vector_type(8)))  float  v8f;

#define NPT 1000000     // points
#define PCL 8192        // polylines
#define HID 64

// ---------------- workspace layout (bytes) ----------------
static constexpr size_t alg(size_t x) { return (x + 255) & ~(size_t)255; }
// bf16 weight tables, [64][Kpad] each (transposed, zero padded)
static constexpr size_t WT_ELEMS   = 69632;                    // bf16 elements
static constexpr size_t OFF_WT     = 0;
static constexpr size_t SZ_XBUF    = (size_t)NPT * 64 * 2;     // bf16 activations
static constexpr size_t OFF_BUFA   = alg(OFF_WT + WT_ELEMS * 2);
static constexpr size_t OFF_BUFB   = alg(OFF_BUFA + SZ_XBUF);
static constexpr size_t SZ_XM      = (size_t)PCL * 64 * 4;     // f32 per-cluster max
static constexpr size_t OFF_XM1    = alg(OFF_BUFB + SZ_XBUF);
static constexpr size_t OFF_XM2    = alg(OFF_XM1 + SZ_XM);
static constexpr size_t OFF_XM3    = alg(OFF_XM2 + SZ_XM);
static constexpr size_t OFF_XM4    = alg(OFF_XM3 + SZ_XM);
static constexpr size_t OFF_COUNTS = alg(OFF_XM4 + SZ_XM);
static constexpr size_t OFF_OFFS   = alg(OFF_COUNTS + PCL * 4);
static constexpr size_t OFF_CURSOR = alg(OFF_OFFS + (PCL + 1) * 4);
static constexpr size_t OFF_ROWS   = alg(OFF_CURSOR + PCL * 4);
static constexpr size_t OFF_G      = alg(OFF_ROWS + (size_t)NPT * 4);  // bf16 [P][96]
static constexpr size_t OFF_K      = alg(OFF_G + (size_t)PCL * 96 * 2);
static constexpr size_t OFF_V      = alg(OFF_K + (size_t)PCL * 64 * 2);
static constexpr size_t OFF_GF0    = alg(OFF_V + (size_t)PCL * 64 * 2);

// weight-table element offsets (bf16 elements inside WT region)
//  idx: 0 s1W1  1 s1W2  2 s1Wsc  3 s2W1  4 s2W2  5 s2Wsc
//       6 s3W1  7 s3W2  8 s3Wsc  9 lin   10 Wk   11 Wv
#define WT_S1W1  0
#define WT_S1W2  2048
#define WT_S1WSC 6144
#define WT_S2W1  8192
#define WT_S2W2  16384
#define WT_S2WSC 20480
#define WT_S3W1  28672
#define WT_S3W2  36864
#define WT_S3WSC 40960
#define WT_LIN   49152
#define WT_WK    57344
#define WT_WV    63488

// ---------------- device helpers ----------------
static __device__ __forceinline__ float bf2f(bf16 x) { return (float)x; }
static __device__ __forceinline__ bf16  f2bf(float x) { return (bf16)x; }

static __device__ __forceinline__ v8f wmma_bf16(v16bf a, v16bf b, v8f c) {
    return __builtin_amdgcn_wmma_f32_16x16x32_bf16(false, a, false, b, (short)0, c,
                                                   false, false);
}

// A-fragment (16x32 bf16) from row-major LDS tile with leading dim `ld` elems.
// Lanes 0-15 / 16-31 both hold M=0..15; half h holds K = h*16..h*16+15
// (contiguous, matching load_B).
static __device__ __forceinline__ v16bf load_A(const bf16* sm, int ld, int kbase) {
    int lane = threadIdx.x & 31;
    int r = lane & 15, h = lane >> 4;
    const bf16* p0 = sm + r * ld + kbase + h * 16;
    v16bf a;
#pragma unroll
    for (int i = 0; i < 16; ++i) a[i] = p0[i];
    return a;
}

// B-fragment (32x16 bf16) from transposed weights wt[64][kpad] (row = out idx).
// Lane holds column n = lane&15; half h holds K = h*16 .. h*16+15 (contiguous).
static __device__ __forceinline__ v16bf load_B(const bf16* wt, int kpad, int ntile,
                                               int kbase) {
    int lane = threadIdx.x & 31;
    int n = lane & 15, h = lane >> 4;
    const bf16* p = wt + (size_t)(ntile * 16 + n) * kpad + kbase + h * 16;
    v16bf b;
#pragma unroll
    for (int i = 0; i < 16; ++i) b[i] = p[i];
    return b;
}

// 16x64 = A[16xK] @ Wt^T, bias-initialized accumulators, dump f32 to LDS [16][dld].
template <int KS>
static __device__ __forceinline__ void mm_tile(const bf16* A, int lda, const bf16* wt,
                                               int kpad, const float* bias, float* dump,
                                               int dld) {
    int lane = threadIdx.x & 31;
    int n = lane & 15, h = lane >> 4;
    v16bf af[KS];
#pragma unroll
    for (int kb = 0; kb < KS; ++kb) af[kb] = load_A(A, lda, kb * 32);
#pragma unroll
    for (int nt = 0; nt < 4; ++nt) {
        float bv = bias[nt * 16 + n];
        v8f c;
#pragma unroll
        for (int r = 0; r < 8; ++r) c[r] = bv;
#pragma unroll
        for (int kb = 0; kb < KS; ++kb)
            c = wmma_bf16(af[kb], load_B(wt, kpad, nt, kb * 32), c);
#pragma unroll
        for (int r = 0; r < 8; ++r) dump[(r + 8 * h) * dld + nt * 16 + n] = c[r];
    }
}

// LayerNorm(+ReLU) over 64 cols of a 16-row f32 dump -> bf16 LDS tile [16][old].
// Lane (r,h) owns row r, cols h*32..h*32+31; stats combined with shfl_xor(16).
static __device__ __forceinline__ void ln_relu_store(const float* dump, int dld,
                                                     const float* gg, const float* bb,
                                                     bf16* outsm, int old) {
    int lane = threadIdx.x & 31;
    int r = lane & 15, h = lane >> 4;
    const float* rowp = dump + r * dld + h * 32;
    float s1 = 0.f, s2 = 0.f;
#pragma unroll
    for (int i = 0; i < 32; ++i) { float v = rowp[i]; s1 += v; s2 += v * v; }
    s1 += __shfl_xor(s1, 16); s2 += __shfl_xor(s2, 16);
    float mean = s1 * (1.f / 64.f);
    float var  = s2 * (1.f / 64.f) - mean * mean;
    float rstd = rsqrtf(var + 1e-5f);
    bf16* orow = outsm + r * old + h * 32;
#pragma unroll
    for (int i = 0; i < 32; ++i) {
        int c = h * 32 + i;
        float y = (rowp[i] - mean) * rstd * gg[c] + bb[c];
        orow[i] = f2bf(fmaxf(y, 0.f));
    }
}

// relu(LN(dA)+LN(dB)) -> packed bf16 global store (row stride 64).
static __device__ __forceinline__ void ln2_combine_store(
    const float* dA, const float* dB, int dld, const float* ga, const float* na,
    const float* gb, const float* nb, bf16* gout) {
    int lane = threadIdx.x & 31;
    int r = lane & 15, h = lane >> 4;
    const float* pa = dA + r * dld + h * 32;
    const float* pb = dB + r * dld + h * 32;
    float sa1 = 0.f, sa2 = 0.f, sb1 = 0.f, sb2 = 0.f;
#pragma unroll
    for (int i = 0; i < 32; ++i) {
        float va = pa[i]; sa1 += va; sa2 += va * va;
        float vb = pb[i]; sb1 += vb; sb2 += vb * vb;
    }
    sa1 += __shfl_xor(sa1, 16); sa2 += __shfl_xor(sa2, 16);
    sb1 += __shfl_xor(sb1, 16); sb2 += __shfl_xor(sb2, 16);
    float ma = sa1 * (1.f / 64.f), va_ = sa2 * (1.f / 64.f) - ma * ma;
    float mb = sb1 * (1.f / 64.f), vb_ = sb2 * (1.f / 64.f) - mb * mb;
    float ra = rsqrtf(va_ + 1e-5f), rb = rsqrtf(vb_ + 1e-5f);
    alignas(16) bf16 tmp[32];
#pragma unroll
    for (int i = 0; i < 32; ++i) {
        int c = h * 32 + i;
        float y = (pa[i] - ma) * ra * ga[c] + na[c] + (pb[i] - mb) * rb * gb[c] + nb[c];
        tmp[i] = f2bf(fmaxf(y, 0.f));
    }
    uint4* dst = (uint4*)(gout + (size_t)r * 64 + h * 32);
    const uint4* s = (const uint4*)tmp;
#pragma unroll
    for (int j = 0; j < 4; ++j) dst[j] = s[j];
}

// f32 dump -> packed bf16 global store (no LN).
static __device__ __forceinline__ void dump_store_bf16(const float* d, int dld,
                                                       bf16* gout) {
    int lane = threadIdx.x & 31;
    int r = lane & 15, h = lane >> 4;
    const float* p = d + r * dld + h * 32;
    alignas(16) bf16 tmp[32];
#pragma unroll
    for (int i = 0; i < 32; ++i) tmp[i] = f2bf(p[i]);
    uint4* dst = (uint4*)(gout + (size_t)r * 64 + h * 32);
    const uint4* s = (const uint4*)tmp;
#pragma unroll
    for (int j = 0; j < 4; ++j) dst[j] = s[j];
}

// ---------------- weight prep: f32 [K][64] -> bf16 [64][Kpad] ----------------
struct PrepArgs { const float* src[12]; };

__global__ __launch_bounds__(256) void prep_kernel(PrepArgs args, bf16* wt) {
    const int  K[12]  = {6, 64, 6, 128, 64, 128, 128, 64, 128, 128, 66, 66};
    const int  KP[12] = {32, 64, 32, 128, 64, 128, 128, 64, 128, 128, 96, 96};
    const int  OF[12] = {WT_S1W1, WT_S1W2, WT_S1WSC, WT_S2W1, WT_S2W2, WT_S2WSC,
                         WT_S3W1, WT_S3W2, WT_S3WSC, WT_LIN, WT_WK, WT_WV};
    int b = blockIdx.x;
    const float* W = args.src[b];
    int k = K[b], kp = KP[b];
    bf16* dst = wt + OF[b];
    for (int idx = threadIdx.x; idx < 64 * kp; idx += blockDim.x) {
        int n = idx / kp, kk = idx % kp;
        dst[idx] = f2bf(kk < k ? W[kk * 64 + n] : 0.f);
    }
}

// ---------------- CSR build ----------------
__global__ void zero_counts_kernel(int* counts) {
    int i = blockIdx.x * blockDim.x + threadIdx.x;
    if (i < PCL) counts[i] = 0;
}
__global__ void count_kernel(const int* __restrict__ cluster, int* __restrict__ counts) {
    int i = blockIdx.x * blockDim.x + threadIdx.x;
    if (i < NPT) atomicAdd(&counts[cluster[i]], 1);
}
__global__ __launch_bounds__(256) void scan_kernel(const int* __restrict__ counts,
                                                   int* __restrict__ offs,
                                                   int* __restrict__ cursor) {
    __shared__ int part[256];
    int t = threadIdx.x;
    int local[32];
    int s = 0, base = t * 32;
#pragma unroll
    for (int i = 0; i < 32; ++i) { local[i] = counts[base + i]; s += local[i]; }
    part[t] = s;
    __syncthreads();
    for (int off = 1; off < 256; off <<= 1) {
        int v = (t >= off) ? part[t - off] : 0;
        __syncthreads();
        part[t] += v;
        __syncthreads();
    }
    int run = part[t] - s;  // exclusive prefix of this 32-chunk
#pragma unroll
    for (int i = 0; i < 32; ++i) {
        offs[base + i] = run; cursor[base + i] = run; run += local[i];
    }
    if (t == 255) offs[PCL] = run;
}
__global__ void scatter_kernel(const int* __restrict__ cluster, int* __restrict__ cursor,
                               int* __restrict__ rows) {
    int i = blockIdx.x * blockDim.x + threadIdx.x;
    if (i < NPT) { int pos = atomicAdd(&cursor[cluster[i]], 1); rows[pos] = i; }
}

// ---------------- gather seg-max ----------------
__global__ __launch_bounds__(64) void segmax_kernel(const bf16* __restrict__ x,
                                                    const int* __restrict__ offs,
                                                    const int* __restrict__ rows,
                                                    float* __restrict__ xm) {
    int p = blockIdx.x, t = threadIdx.x;
    int beg = offs[p], end = offs[p + 1];
    float acc = -INFINITY;
    for (int i = beg; i < end; ++i)
        acc = fmaxf(acc, bf2f(x[(size_t)rows[i] * 64 + t]));
    if (!(acc > -INFINITY)) acc = 0.f;  // empty segment -> 0 (matches reference)
    xm[(size_t)p * 64 + t] = acc;
}

// ---------------- stage 1: x0[N,6] -> mlp(6->64) -> x1 (bf16) ----------------
__global__ __launch_bounds__(128) void stage1_kernel(
    const float* __restrict__ x0, const bf16* __restrict__ wt,
    const float* b1, const float* g1, const float* n1,
    const float* b2, const float* g2, const float* n2,
    const float* bsc, const float* gsc, const float* nsc,
    bf16* __restrict__ xout) {
    __shared__ bf16  sm_x[4][16][40];
    __shared__ bf16  sm_h[4][16][72];
    __shared__ float sm_a[4][16][68];
    __shared__ float sm_b[4][16][68];
    int wave = threadIdx.x >> 5;
    int tile0 = blockIdx.x * 4;
    if (threadIdx.x < 64) {
        int rloc = threadIdx.x, w = rloc >> 4, rr = rloc & 15;
        size_t grow = (size_t)tile0 * 16 + rloc;
        const float* src = x0 + grow * 6;
#pragma unroll
        for (int i = 0; i < 6; ++i) sm_x[w][rr][i] = f2bf(src[i]);
#pragma unroll
        for (int i = 6; i < 40; ++i) sm_x[w][rr][i] = f2bf(0.f);
    }
    __syncthreads();
    const bf16* A = &sm_x[wave][0][0];
    bf16* H = &sm_h[wave][0][0];
    float* dA = &sm_a[wave][0][0];
    float* dB = &sm_b[wave][0][0];
    mm_tile<1>(A, 40, wt + WT_S1W1, 32, b1, dA, 68);      // h_pre
    ln_relu_store(dA, 68, g1, n1, H, 72);                 // h
    mm_tile<1>(A, 40, wt + WT_S1WSC, 32, bsc, dB, 68);    // shortcut pre
    mm_tile<2>(H, 72, wt + WT_S1W2, 64, b2, dA, 68);      // o_pre
    ln2_combine_store(dA, dB, 68, g2, n2, gsc, nsc,
                      xout + (size_t)(tile0 + wave) * 16 * 64);
}

// ---------------- stages 2/3: [x_prev | xm_prev[c]] (128) -> mlp -> out ------
__global__ __launch_bounds__(128) void stage_mid_kernel(
    const bf16* __restrict__ xin, const float* __restrict__ xmprev,
    const int* __restrict__ cluster, const bf16* __restrict__ w1t,
    const bf16* __restrict__ w2t, const bf16* __restrict__ wsct,
    const float* b1, const float* g1, const float* n1,
    const float* b2, const float* g2, const float* n2,
    const float* bsc, const float* gsc, const float* nsc,
    bf16* __restrict__ xout) {
    __shared__ bf16  sm_x[4][16][136];
    __shared__ bf16  sm_h[4][16][72];
    __shared__ float sm_a[4][16][68];
    __shared__ float sm_b[4][16][68];
    int wave = threadIdx.x >> 5;
    int tile0 = blockIdx.x * 4;
    {
        int t = threadIdx.x;
        int rloc = t >> 1, part = t & 1;
        int w = rloc >> 4, rr = rloc & 15;
        size_t grow = (size_t)tile0 * 16 + rloc;
        if (part == 0) {
            const bf16* src = xin + grow * 64;
#pragma unroll
            for (int i = 0; i < 64; ++i) sm_x[w][rr][i] = src[i];
        } else {
            const float* src = xmprev + (size_t)cluster[grow] * 64;
#pragma unroll
            for (int i = 0; i < 64; ++i) sm_x[w][rr][64 + i] = f2bf(src[i]);
#pragma unroll
            for (int i = 128; i < 136; ++i) sm_x[w][rr][i] = f2bf(0.f);
        }
    }
    __syncthreads();
    const bf16* A = &sm_x[wave][0][0];
    bf16* H = &sm_h[wave][0][0];
    float* dA = &sm_a[wave][0][0];
    float* dB = &sm_b[wave][0][0];
    mm_tile<4>(A, 136, w1t, 128, b1, dA, 68);
    ln_relu_store(dA, 68, g1, n1, H, 72);
    mm_tile<4>(A, 136, wsct, 128, bsc, dB, 68);
    mm_tile<2>(H, 72, w2t, 64, b2, dA, 68);
    ln2_combine_store(dA, dB, 68, g2, n2, gsc, nsc,
                      xout + (size_t)(tile0 + wave) * 16 * 64);
}

// ---------------- sub_lin: [x3 | xm3[c]] @ W [128,64] + b -> x4 (bf16) -------
__global__ __launch_bounds__(128) void stage_lin_kernel(
    const bf16* __restrict__ xin, const float* __restrict__ xmprev,
    const int* __restrict__ cluster, const bf16* __restrict__ wlint,
    const float* blin, bf16* __restrict__ xout) {
    __shared__ bf16  sm_x[4][16][136];
    __shared__ float sm_a[4][16][68];
    int wave = threadIdx.x >> 5;
    int tile0 = blockIdx.x * 4;
    {
        int t = threadIdx.x;
        int rloc = t >> 1, part = t & 1;
        int w = rloc >> 4, rr = rloc & 15;
        size_t grow = (size_t)tile0 * 16 + rloc;
        if (part == 0) {
            const bf16* src = xin + grow * 64;
#pragma unroll
            for (int i = 0; i < 64; ++i) sm_x[w][rr][i] = src[i];
        } else {
            const float* src = xmprev + (size_t)cluster[grow] * 64;
#pragma unroll
            for (int i = 0; i < 64; ++i) sm_x[w][rr][64 + i] = f2bf(src[i]);
#pragma unroll
            for (int i = 128; i < 136; ++i) sm_x[w][rr][i] = f2bf(0.f);
        }
    }
    __syncthreads();
    float* dA = &sm_a[wave][0][0];
    mm_tile<4>(&sm_x[wave][0][0], 136, wlint, 128, blin, dA, 68);
    dump_store_bf16(dA, 68, xout + (size_t)(tile0 + wave) * 16 * 64);
}

// ---------------- build g = [normalize(xm4) | id_emb | zero-pad] bf16 [P][96]
__global__ __launch_bounds__(64) void build_g_kernel(const float* __restrict__ xm4,
                                                     const float* __restrict__ id_emb,
                                                     bf16* __restrict__ g) {
    __shared__ float red[64];
    int p = blockIdx.x, t = threadIdx.x;
    float v = xm4[(size_t)p * 64 + t];
    red[t] = v * v;
    __syncthreads();
    for (int s = 32; s > 0; s >>= 1) { if (t < s) red[t] += red[t + s]; __syncthreads(); }
    float inv = 1.f / fmaxf(sqrtf(red[0]), 1e-12f);
    g[(size_t)p * 96 + t] = f2bf(v * inv);
    if (t < 2)  g[(size_t)p * 96 + 64 + t] = f2bf(id_emb[p * 2 + t]);
    if (t < 30) g[(size_t)p * 96 + 66 + t] = f2bf(0.f);
}

// ---------------- K/V projections: g[P,96] @ W[96pad,64] (WMMA) -------------
__global__ __launch_bounds__(128) void qkv_kernel(const bf16* __restrict__ g,
                                                  const bf16* __restrict__ wkt,
                                                  const bf16* __restrict__ wvt,
                                                  const float* bk, const float* bv,
                                                  bf16* __restrict__ kout,
                                                  bf16* __restrict__ vout) {
    __shared__ bf16  sm_g[4][16][104];
    __shared__ float sm_d[4][16][68];
    int wave = threadIdx.x >> 5;
    int tile0 = blockIdx.x * 4;
    {
        int t = threadIdx.x;
        int rloc = t >> 1, part = t & 1;
        int w = rloc >> 4, rr = rloc & 15;
        const bf16* src = g + (size_t)(tile0 * 16 + rloc) * 96;
        if (part == 0) {
#pragma unroll
            for (int i = 0; i < 48; ++i) sm_g[w][rr][i] = src[i];
        } else {
#pragma unroll
            for (int i = 0; i < 48; ++i) sm_g[w][rr][48 + i] = src[48 + i];
#pragma unroll
            for (int i = 96; i < 104; ++i) sm_g[w][rr][i] = f2bf(0.f);
        }
    }
    __syncthreads();
    const bf16* A = &sm_g[wave][0][0];
    float* d = &sm_d[wave][0][0];
    mm_tile<3>(A, 104, wkt, 96, bk, d, 68);
    dump_store_bf16(d, 68, kout + (size_t)(tile0 + wave) * 16 * 64);
    mm_tile<3>(A, 104, wvt, 96, bv, d, 68);
    dump_store_bf16(d, 68, vout + (size_t)(tile0 + wave) * 16 * 64);
}

// ---------------- single-query attention (only gf[0] is consumed) -----------
__global__ __launch_bounds__(1024) void attn_kernel(const bf16* __restrict__ g,
                                                    const float* __restrict__ Wq,
                                                    const float* __restrict__ bq,
                                                    const bf16* __restrict__ kbuf,
                                                    const bf16* __restrict__ vbuf,
                                                    float* __restrict__ gf0) {
    __shared__ float q0[64];
    __shared__ float sc[PCL];
    __shared__ float red[1024];
    int t = threadIdx.x;
    if (t < 64) {
        float acc = bq[t];
        for (int i = 0; i < 66; ++i) acc += bf2f(g[i]) * Wq[i * 64 + t];
        q0[t] = acc;
    }
    __syncthreads();
    float lmax = -INFINITY;
    for (int j = t; j < PCL; j += 1024) {
        const bf16* kr = kbuf + (size_t)j * 64;
        float s = 0.f;
#pragma unroll
        for (int d = 0; d < 64; ++d) s += q0[d] * bf2f(kr[d]);
        s *= 0.125f;  // 1/sqrt(64)
        sc[j] = s;
        lmax = fmaxf(lmax, s);
    }
    red[t] = lmax;
    __syncthreads();
    for (int s = 512; s > 0; s >>= 1) {
        if (t < s) red[t] = fmaxf(red[t], red[t + s]);
        __syncthreads();
    }
    float mx = red[0];
    __syncthreads();
    float lsum = 0.f;
    for (int j = t; j < PCL; j += 1024) { float e = expf(sc[j] - mx); sc[j] = e; lsum += e; }
    red[t] = lsum;
    __syncthreads();
    for (int s = 512; s > 0; s >>= 1) {
        if (t < s) red[t] += red[t + s];
        __syncthreads();
    }
    float inv = 1.f / red[0];
    __syncthreads();
    int d = t & 63, grp = t >> 6;  // 16 groups x 64 dims
    float acc = 0.f;
    for (int j = grp; j < PCL; j += 16) acc += sc[j] * bf2f(vbuf[(size_t)j * 64 + d]);
    red[t] = acc;
    __syncthreads();
    if (t < 64) {
        float a = 0.f;
#pragma unroll
        for (int gi = 0; gi < 16; ++gi) a += red[gi * 64 + t];
        gf0[t] = a * inv;
    }
}

// ---------------- pred head (1 row): mlp_apply + linear + cumsum ------------
static __device__ __forceinline__ float ln64(float v, float gg, float bb, float* red) {
    int t = threadIdx.x;
    red[t] = v;
    __syncthreads();
    for (int s = 32; s > 0; s >>= 1) { if (t < s) red[t] += red[t + s]; __syncthreads(); }
    float mean = red[0] * (1.f / 64.f);
    __syncthreads();
    red[t] = (v - mean) * (v - mean);
    __syncthreads();
    for (int s = 32; s > 0; s >>= 1) { if (t < s) red[t] += red[t + s]; __syncthreads(); }
    float var = red[0] * (1.f / 64.f);
    __syncthreads();
    return (v - mean) * rsqrtf(var + 1e-5f) * gg + bb;
}

__global__ __launch_bounds__(64) void pred_kernel(
    const float* __restrict__ gf0, const float* W1, const float* b1, const float* g1,
    const float* n1, const float* W2, const float* b2, const float* g2, const float* n2,
    const float* pW, const float* pb, float* __restrict__ out) {
    __shared__ float x[64], h[64], hf[64], red[64], predv[60];
    int t = threadIdx.x;
    x[t] = gf0[t];
    __syncthreads();
    float a = b1[t];
    for (int i = 0; i < 64; ++i) a += x[i] * W1[i * 64 + t];
    float hv = fmaxf(ln64(a, g1[t], n1[t], red), 0.f);
    h[t] = hv;
    __syncthreads();
    float b = b2[t];
    for (int i = 0; i < 64; ++i) b += h[i] * W2[i * 64 + t];
    float o = ln64(b, g2[t], n2[t], red);
    hf[t] = fmaxf(o + x[t], 0.f);  // identity shortcut (cin==cout)
    __syncthreads();
    if (t < 60) {
        float p = pb[t];
        for (int i = 0; i < 64; ++i) p += hf[i] * pW[i * 60 + t];
        predv[t] = p;
    }
    __syncthreads();
    if (t < 60) {
        int step = t >> 1, c = t & 1;
        float s = 0.f;
        for (int k = 0; k <= step; ++k) s += predv[k * 2 + c];
        out[t] = s;
    }
}

// ---------------- host launcher ----------------
extern "C" void kernel_launch(void* const* d_in, const int* in_sizes, int n_in,
                              void* d_out, int out_size, void* d_ws, size_t ws_size,
                              hipStream_t stream) {
    (void)in_sizes; (void)n_in; (void)out_size; (void)ws_size;
    const float* x0      = (const float*)d_in[0];
    const int*   cluster = (const int*)d_in[1];
    const float* id_emb  = (const float*)d_in[2];
    // params (see index map in analysis)
    auto F = [&](int i) { return (const float*)d_in[i]; };

    char* ws = (char*)d_ws;
    bf16* wt     = (bf16*)(ws + OFF_WT);
    bf16* bufA   = (bf16*)(ws + OFF_BUFA);
    bf16* bufB   = (bf16*)(ws + OFF_BUFB);
    float* xm1   = (float*)(ws + OFF_XM1);
    float* xm2   = (float*)(ws + OFF_XM2);
    float* xm3   = (float*)(ws + OFF_XM3);
    float* xm4   = (float*)(ws + OFF_XM4);
    int* counts  = (int*)(ws + OFF_COUNTS);
    int* offs    = (int*)(ws + OFF_OFFS);
    int* cursor  = (int*)(ws + OFF_CURSOR);
    int* rows    = (int*)(ws + OFF_ROWS);
    bf16* gbuf   = (bf16*)(ws + OFF_G);
    bf16* kbuf   = (bf16*)(ws + OFF_K);
    bf16* vbuf   = (bf16*)(ws + OFF_V);
    float* gf0   = (float*)(ws + OFF_GF0);

    // weight prep
    PrepArgs pa;
    pa.src[0] = F(4);  pa.src[1] = F(8);  pa.src[2] = F(12);
    pa.src[3] = F(16); pa.src[4] = F(20); pa.src[5] = F(24);
    pa.src[6] = F(28); pa.src[7] = F(32); pa.src[8] = F(36);
    pa.src[9] = F(40); pa.src[10] = F(44); pa.src[11] = F(46);
    prep_kernel<<<12, 256, 0, stream>>>(pa, wt);

    // CSR build (cluster -> member rows)
    zero_counts_kernel<<<(PCL + 255) / 256, 256, 0, stream>>>(counts);
    count_kernel<<<(NPT + 255) / 256, 256, 0, stream>>>(cluster, counts);
    scan_kernel<<<1, 256, 0, stream>>>(counts, offs, cursor);
    scatter_kernel<<<(NPT + 255) / 256, 256, 0, stream>>>(cluster, cursor, rows);

    const int SGRID = NPT / 64;  // 15625 blocks x (4 waves x 16 rows)
    // stage 1
    stage1_kernel<<<SGRID, 128, 0, stream>>>(x0, wt, F(5), F(6), F(7), F(9), F(10),
                                             F(11), F(13), F(14), F(15), bufA);
    segmax_kernel<<<PCL, 64, 0, stream>>>(bufA, offs, rows, xm1);
    // stage 2
    stage_mid_kernel<<<SGRID, 128, 0, stream>>>(
        bufA, xm1, cluster, wt + WT_S2W1, wt + WT_S2W2, wt + WT_S2WSC, F(17), F(18),
        F(19), F(21), F(22), F(23), F(25), F(26), F(27), bufB);
    segmax_kernel<<<PCL, 64, 0, stream>>>(bufB, offs, rows, xm2);
    // stage 3
    stage_mid_kernel<<<SGRID, 128, 0, stream>>>(
        bufB, xm2, cluster, wt + WT_S3W1, wt + WT_S3W2, wt + WT_S3WSC, F(29), F(30),
        F(31), F(33), F(34), F(35), F(37), F(38), F(39), bufA);
    segmax_kernel<<<PCL, 64, 0, stream>>>(bufA, offs, rows, xm3);
    // sub_lin
    stage_lin_kernel<<<SGRID, 128, 0, stream>>>(bufA, xm3, cluster, wt + WT_LIN, F(41),
                                                bufB);
    segmax_kernel<<<PCL, 64, 0, stream>>>(bufB, offs, rows, xm4);

    // global graph (single query)
    build_g_kernel<<<PCL, 64, 0, stream>>>(xm4, id_emb, gbuf);
    qkv_kernel<<<PCL / 64, 128, 0, stream>>>(gbuf, wt + WT_WK, wt + WT_WV, F(45), F(47),
                                             kbuf, vbuf);
    attn_kernel<<<1, 1024, 0, stream>>>(gbuf, F(42), F(43), kbuf, vbuf, gf0);

    // pred head
    pred_kernel<<<1, 64, 0, stream>>>(gf0, F(48), F(49), F(50), F(51), F(52), F(53),
                                      F(54), F(55), F(56), F(57), (float*)d_out);
}
